// EvolutionaryFeatureExtractor_25142738550966
// MI455X (gfx1250) — 8-bit matrix operand order repaired and checked by reading, compiled, not run on hardware
//
#include <hip/hip_runtime.h>
#include <stdint.h>


typedef __attribute__((ext_vector_type(16))) int   v16i;
typedef __attribute__((ext_vector_type(8)))  float v8f;

#define AA      21
#define NSEQ    512
#define LPOS    256
#define PPAD    32
#define MSA_ROW (LPOS * AA)
#define POS_DW  (PPAD * NSEQ / 4)
#define ROW_DW  (NSEQ / 4)
#define EPSF    1e-9f
#define LOG2A   4.3923174227787602f

__device__ __forceinline__ float wave_sum(float x) {
#pragma unroll
  for (int o = 16; o > 0; o >>= 1) x += __shfl_xor(x, o, 32);
  return x;
}

__device__ __forceinline__ float block_sum_256(float x) {
  __shared__ float red[8];
  float w = wave_sum(x);
  int lane = threadIdx.x & 31, wv = threadIdx.x >> 5;
  if (lane == 0) red[wv] = w;
  __syncthreads();
  float r = (threadIdx.x < 8) ? red[threadIdx.x] : 0.0f;
  if (wv == 0) r = wave_sum(r);
  return r;
}

__device__ __forceinline__ float plogp(float c) {
  float p = c * (1.0f / (float)NSEQ) + EPSF;
  return p * __log2f(p);
}

__device__ __forceinline__ int kperm(int l) {
  return (((l >> 3) & 1) << 6) | (((l >> 6) & 1) << 5) |
         (((l >> 4) & 3) << 3) | (((l >> 2) & 1) << 2) | (l & 3);
}

__global__ void pack_a_kernel(const float* __restrict__ msa,
                              uint32_t* __restrict__ ya) {
  int idx = blockIdx.x * blockDim.x + threadIdx.x;
  int k4 = idx & 127;
  int p  = (idx >> 7) & 31;
  int i  = idx >> 12;
  uint32_t w = 0u;
  if (p < AA) {
#pragma unroll
    for (int t = 0; t < 4; ++t) {
      int k = k4 * 4 + t;
      float x = msa[k * MSA_ROW + i * AA + p];
      w |= (x > 0.5f ? 0x38u : 0x00u) << (8 * t);
    }
  }
  ya[idx] = w;
}

__global__ void pack_b_kernel(const float* __restrict__ msa,
                              uint32_t* __restrict__ yb) {
  int idx = blockIdx.x * blockDim.x + threadIdx.x;
  int q  = idx & 127;
  int b  = (idx >> 7) & 31;
  int j  = idx >> 12;
  int rc = q & 15;
  int lh = (q >> 4) & 1;
  int kc = q >> 5;
  uint32_t w = 0u;
  if (b < AA) {
#pragma unroll
    for (int t = 0; t < 4; ++t) {
      int kp  = (rc >> 2) * 32 + lh * 16 + (rc & 3) * 4 + t;
      int seq = kc * 128 + lh * 64 + rc * 4 + t;
      float x = msa[seq * MSA_ROW + j * AA + b];
      w |= (x > 0.5f ? 0x38u : 0x00u) << (8 * t);
    }
  }
  yb[idx] = w;
}

__global__ void stats_kernel(const float* __restrict__ msa,
                             float* __restrict__ out,
                             float* __restrict__ Tterm) {
  int i = blockIdx.x;
  int a = threadIdx.x;
  float cnt = 0.0f;
  if (a < AA) {
    const float* col = msa + i * AA + a;
    for (int k = 0; k < NSEQ; ++k) cnt += col[k * MSA_ROW];
  }
  float mean = cnt * (1.0f / (float)NSEQ);

  float freqP = (a < AA) ? (mean + 0.001f) : 0.0f;
  float sumF  = wave_sum(freqP);
  if (a < AA) out[i * AA + a] = __log2f(freqP / sumF * (float)AA);

  float f2 = mean + EPSF;
  float fl = (a < AA) ? f2 * __log2f(f2) : 0.0f;
  float sl = wave_sum(fl);
  if (a == 0) out[LPOS * AA + i] = 1.0f + sl * (1.0f / LOG2A);

  float tc = (a < AA) ? (mean + (float)AA * EPSF) * __log2f(mean + EPSF) : 0.0f;
  float ts = wave_sum(tc);
  if (a == 0) Tterm[i] = ts;
}

__global__ void __launch_bounds__(256) mi_kernel(
    const uint32_t* __restrict__ ya, const uint32_t* __restrict__ yb,
    const float* __restrict__ Tterm, float* __restrict__ mi) {
  int lane = threadIdx.x & 31;
  int wave = threadIdx.x >> 5;
  int pair = blockIdx.x * 8 + wave;
  int i = pair >> 8;
  int j = pair & 255;
  int n  = lane & 15;
  int lh = lane >> 4;

  const uint32_t* pa0 = ya + i * POS_DW + n * ROW_DW + lh * 16;
  const uint32_t* pa1 = pa0 + 16 * ROW_DW;
  const uint32_t* pb0 = yb + j * POS_DW + n * ROW_DW + lh * 16;
  const uint32_t* pb1 = pb0 + 16 * ROW_DW;

  v8f c00 = {}, c10 = {}, c01 = {}, c11 = {};
#pragma unroll 1
  for (int kc = 0; kc < 4; ++kc) {
    v16i a0 = *(const v16i*)pa0; pa0 += 32;
    v16i a1 = *(const v16i*)pa1; pa1 += 32;
    v16i b0 = *(const v16i*)pb0; pb0 += 32;
    v16i b1 = *(const v16i*)pb1; pb1 += 32;
    c00 = __builtin_amdgcn_wmma_f32_16x16x128_fp8_fp8(a0, b0, (short)0, c00, false, false);
    c10 = __builtin_amdgcn_wmma_f32_16x16x128_fp8_fp8(a1, b0, (short)0, c10, false, false);
    c01 = __builtin_amdgcn_wmma_f32_16x16x128_fp8_fp8(a0, b1, (short)0, c01, false, false);
    c11 = __builtin_amdgcn_wmma_f32_16x16x128_fp8_fp8(a1, b1, (short)0, c11, false, false);
  }

  float s = 0.0f;
#pragma unroll
  for (int v = 0; v < 8; ++v) s += plogp(c00[v]);
  if (n < 5) {
#pragma unroll
    for (int v = 0; v < 8; ++v) s += plogp(c01[v]);
  }
  if (lh == 0) {
#pragma unroll
    for (int v = 0; v < 5; ++v) s += plogp(c10[v]);
    if (n < 5) {
#pragma unroll
      for (int v = 0; v < 5; ++v) s += plogp(c11[v]);
    }
  }
  s = wave_sum(s);
  if (lane == 0)
    mi[i * LPOS + j] = (i == j) ? 0.0f : (s - Tterm[i] - Tterm[j]);
}

__global__ void apc_init(float* __restrict__ colsum, float* __restrict__ total) {
  int t = threadIdx.x;
  colsum[t] = 0.0f;
  if (t == 0) total[0] = 0.0f;
}

__global__ void apc_sums(const float* __restrict__ mi,
                         float* __restrict__ rowsum,
                         float* __restrict__ colsum,
                         float* __restrict__ total) {
  int i = blockIdx.x, t = threadIdx.x;
  float v = mi[i * LPOS + t];
  atomicAdd(&colsum[t], v);
  float s = block_sum_256(v);
  if (t == 0) { rowsum[i] = s; atomicAdd(total, s); }
}

__global__ void apc_apply(const float* __restrict__ rowsum,
                          const float* __restrict__ colsum,
                          const float* __restrict__ total,
                          float* __restrict__ mi) {
  int i = blockIdx.x, j = threadIdx.x;
  float rm = rowsum[i] * (1.0f / (float)LPOS);
  float cm = colsum[j] * (1.0f / (float)LPOS);
  float tm = total[0] * (1.0f / (float)(LPOS * LPOS));
  mi[i * LPOS + j] -= rm * cm / (tm + EPSF);
}

extern "C" void kernel_launch(void* const* d_in, const int* in_sizes, int n_in,
                              void* d_out, int out_size, void* d_ws, size_t ws_size,
                              hipStream_t stream) {
  const float* msa = (const float*)d_in[0];
  float* out = (float*)d_out;
  float* out_mi = out + LPOS * AA + LPOS;

  uint32_t* ya  = (uint32_t*)d_ws;
  uint32_t* yb  = ya + LPOS * POS_DW;
  float* wsf    = (float*)d_ws;
  float* Tterm  = wsf + 2 * LPOS * POS_DW;
  float* rowsum = Tterm + LPOS;
  float* colsum = rowsum + LPOS;
  float* total  = colsum + LPOS;

  pack_a_kernel<<<LPOS * POS_DW / 256, 256, 0, stream>>>(msa, ya);
  pack_b_kernel<<<LPOS * POS_DW / 256, 256, 0, stream>>>(msa, yb);
  stats_kernel<<<LPOS, 32, 0, stream>>>(msa, out, Tterm);
  mi_kernel<<<LPOS * LPOS / 8, 256, 0, stream>>>(ya, yb, Tterm, out_mi);
  apc_init<<<1, 256, 0, stream>>>(colsum, total);
  apc_sums<<<LPOS, 256, 0, stream>>>(out_mi, rowsum, colsum, total);
  apc_apply<<<LPOS, 256, 0, stream>>>(rowsum, colsum, total, out_mi);
}
